// ReadUnit_61753039781974
// MI455X (gfx1250) — matrix operand order repaired and checked by reading, not compiled, not run on hardware
//
#include <hip/hip_runtime.h>
#include <hip/hip_bf16.h>
#include <stdint.h>

// ---------------------------------------------------------------------------
// ReadUnit fused pipeline for MI455X (gfx1250, wave32, WMMA bf16,
// double-buffered async global->LDS copies overlapping the WMMAs)
// ---------------------------------------------------------------------------

#define HID   512
#define KBDIM 512
#define NDIM  2048
#define BDIM  32
#define TDIM  4

typedef __attribute__((ext_vector_type(16))) __bf16 v16bf;
typedef __attribute__((ext_vector_type(8)))  __bf16 bf16x8;
typedef __attribute__((ext_vector_type(8)))  float  v8f;
typedef __attribute__((ext_vector_type(4)))  int    v4i;

// GEMM tiling: block = 256 threads = 8 waves; block tile 64(n) x 256(h);
// wave tile 32(n) x 64(h) = 2x4 WMMA 16x16 tiles; K-step 32.
#define TN 64
#define TH 256
#define TK 32
#define APAD 8
#define ASTR (TK + APAD)   // 40 bf16 -> 80B row stride (16B aligned)

// ---- gfx1250 async global->LDS path (compile-safe fallback) ---------------
#if defined(__has_builtin)
#if __has_builtin(__builtin_amdgcn_global_load_async_to_lds_b128) && \
    __has_builtin(__builtin_amdgcn_s_wait_asynccnt)
#define USE_ASYNC_LDS 1
#endif
#endif

#ifdef USE_ASYNC_LDS
typedef __attribute__((address_space(1))) v4i* as1_v4p;
typedef __attribute__((address_space(3))) v4i* as3_v4p;
// Global generic addresses are numerically identical to AS1; generic LDS
// addresses carry the LDS offset in the low 32 bits (ISA flat-addressing
// rule), so integer reinterprets give the correct hardware addresses.
__device__ __forceinline__ void async_cp16(const void* g, void* l) {
    __builtin_amdgcn_global_load_async_to_lds_b128(
        (as1_v4p)(uintptr_t)g, (as3_v4p)(uint32_t)(uintptr_t)l, 0, 0);
}
__device__ __forceinline__ void async_wait0() {
    __builtin_amdgcn_s_wait_asynccnt(0);
}
#else
__device__ __forceinline__ void async_cp16(const void* g, void* l) {
    *(bf16x8*)l = *(const bf16x8*)g;
}
__device__ __forceinline__ void async_wait0() {}
#endif

__device__ __forceinline__ float elu1(float x) {
    return x > 0.0f ? x : (__expf(x) - 1.0f);
}

__device__ __forceinline__ v8f wmma_bf16(v16bf a, v16bf b, v8f c) {
    return __builtin_amdgcn_wmma_f32_16x16x32_bf16(
        /*neg_a=*/false, a, /*neg_b=*/false, b,
        /*c_mod=*/(short)0, c, /*reuse_a=*/false, /*reuse_b=*/false);
}

union FragU { v16bf v; unsigned int u[8]; };

// A fragment (16x32 bf16), ISA 7.12.2 layout:
// dword j: K0 = 2*j + (j>=4 ? 8 : 0) + 8*(lane>=16)
__device__ __forceinline__ v16bf load_a_frag(const __bf16* As, int row_base, int lane) {
    int m   = lane & 15;
    int hi4 = (lane >> 4) << 2;
    const unsigned int* p = (const unsigned int*)(As + (row_base + m) * ASTR);
    FragU f;
    f.u[0] = p[0 + hi4];  f.u[1] = p[1 + hi4];
    f.u[2] = p[2 + hi4];  f.u[3] = p[3 + hi4];
    f.u[4] = p[8 + hi4];  f.u[5] = p[9 + hi4];
    f.u[6] = p[10 + hi4]; f.u[7] = p[11 + hi4];
    return f.v;
}

// B fragment (32x16 bf16) from weight tile stored [h][k]:
// lane<16: N=lane, K=0..7,16..23 ; lane>=16: N=lane-16, K=8..15,24..31
__device__ __forceinline__ v16bf load_b_frag(const __bf16* Bs, int col_base, int lane) {
    int m   = lane & 15;
    int hi8 = (lane >> 4) << 3;
    const unsigned int* p = (const unsigned int*)(Bs + (col_base + m) * ASTR);
    FragU f;
#pragma unroll
    for (int j = 0; j < 8; ++j) f.u[j] = p[(j & 3) + ((j >> 2) << 3) + (hi8 >> 1)];
    return f.v;
}

// Shared wave-tile compute: 2x4 WMMA on one LDS buffer pair.
__device__ __forceinline__ void wave_compute(const __bf16* Ac, const __bf16* Bc,
                                             int wn, int wh, int lane, v8f acc[2][4]) {
    v16bf bfr[4];
#pragma unroll
    for (int th = 0; th < 4; ++th) bfr[th] = load_b_frag(Bc, wh * 64 + th * 16, lane);
#pragma unroll
    for (int tn = 0; tn < 2; ++tn) {
        v16bf af = load_a_frag(Ac, wn * 32 + tn * 16, lane);
#pragma unroll
        for (int th = 0; th < 4; ++th)
            acc[tn][th] = wmma_bf16(af, bfr[th], acc[tn][th]);
    }
}

// ---------------------------------------------------------------------------
// Small prep kernels
// ---------------------------------------------------------------------------

__global__ void projmem_kernel(const float* __restrict__ memory,
                               const float* __restrict__ masks,
                               const float* __restrict__ W_mem,
                               const float* __restrict__ b_mem,
                               float* __restrict__ pm) {
    int b = blockIdx.y;
    int h = blockIdx.x * blockDim.x + threadIdx.x;
    const float* mrow = memory + ((size_t)(TDIM - 1) * BDIM + b) * HID;
    const float* mk   = masks + (size_t)b * HID;
    const float* wrow = W_mem + (size_t)h * HID;
    float s = b_mem[h];
#pragma unroll 4
    for (int k = 0; k < HID; ++k) s += mrow[k] * mk[k] * wrow[k];
    pm[(size_t)b * HID + h] = s;
}

__global__ void cvt_bf16_kernel(const float* __restrict__ src,
                                __bf16* __restrict__ dst, int n) {
    int i = blockIdx.x * blockDim.x + threadIdx.x;
    if (i < n) dst[i] = (__bf16)src[i];
}

__global__ void zero_f32_kernel(float* __restrict__ p, int n) {
    int i = blockIdx.x * blockDim.x + threadIdx.x;
    if (i < n) p[i] = 0.0f;
}

// kT[b][n][k] = (bf16) know[b][k][n]   (32x32 tile transpose via LDS)
__global__ void transpose_know_kernel(const float* __restrict__ know,
                                      __bf16* __restrict__ kT) {
    __shared__ float t[32][33];
    int n0 = blockIdx.x * 32, k0 = blockIdx.y * 32, b = blockIdx.z;
    int tx = threadIdx.x, ty = threadIdx.y;  // (32, 8)
    const float* src = know + ((size_t)b * KBDIM + k0) * NDIM + n0;
#pragma unroll
    for (int i = 0; i < 4; ++i) t[ty + 8 * i][tx] = src[(size_t)(ty + 8 * i) * NDIM + tx];
    __syncthreads();
    __bf16* dst = kT + ((size_t)b * NDIM + n0) * KBDIM + k0;
#pragma unroll
    for (int i = 0; i < 4; ++i)
        dst[(size_t)(ty + 8 * i) * KBDIM + tx] = (__bf16)t[tx][ty + 8 * i];
}

// ---------------------------------------------------------------------------
// GEMM 1: P = kT @ Wkb^T + b_kb ; also writes G = pm .* P   (K = 512)
// ---------------------------------------------------------------------------
__global__ void __launch_bounds__(256)
gemm_projknow_kernel(const __bf16* __restrict__ kT,
                     const __bf16* __restrict__ Wkb,
                     const float* __restrict__ b_kb,
                     const float* __restrict__ pm,
                     __bf16* __restrict__ P,
                     __bf16* __restrict__ G) {
    __shared__ __attribute__((aligned(16))) __bf16 As[2][TN * ASTR];
    __shared__ __attribute__((aligned(16))) __bf16 Bs[2][TH * ASTR];

    int n0 = blockIdx.x * TN, h0 = blockIdx.y * TH, b = blockIdx.z;
    int tid = threadIdx.x, lane = tid & 31, wave = tid >> 5;
    int wn = wave & 1, wh = wave >> 1;

    int arow = tid >> 2, acol = (tid & 3) * 8;   // 64 rows x 32 cols, 8 bf16/thread
    const __bf16* agp = kT + ((size_t)b * NDIM + n0 + arow) * KBDIM + acol;
    const __bf16* bgp = Wkb + (size_t)(h0 + tid) * KBDIM;  // 256 rows, 1 thread/row

    auto issue = [&](int ks, int buf) {
        const __bf16* as = agp + ks * TK;
        const __bf16* bs = bgp + ks * TK;
        async_cp16(as, &As[buf][arow * ASTR + acol]);
#pragma unroll
        for (int c = 0; c < 4; ++c)
            async_cp16(bs + c * 8, &Bs[buf][tid * ASTR + c * 8]);
        __builtin_prefetch(as + TK, 0, 3);
        __builtin_prefetch(bs + TK, 0, 3);
    };

    v8f acc[2][4] = {};
    constexpr int NSTEP = KBDIM / TK;

    issue(0, 0);
#pragma unroll 2
    for (int ks = 0; ks < NSTEP; ++ks) {
        int cur = ks & 1;
        async_wait0();
        __syncthreads();
        if (ks + 1 < NSTEP) issue(ks + 1, cur ^ 1);
        wave_compute(As[cur], Bs[cur], wn, wh, lane, acc);
    }

    int m = lane & 15, hi = lane >> 4;
#pragma unroll
    for (int tn = 0; tn < 2; ++tn)
#pragma unroll
        for (int th = 0; th < 4; ++th) {
            int hh = h0 + wh * 64 + th * 16 + m;
            float bias = b_kb[hh];
            float pmv  = pm[(size_t)b * HID + hh];
#pragma unroll
            for (int r = 0; r < 8; ++r) {
                int nn = n0 + wn * 32 + tn * 16 + hi * 8 + r;
                size_t idx = ((size_t)b * NDIM + nn) * HID + hh;
                float v = acc[tn][th][r] + bias;
                P[idx] = (__bf16)v;
                G[idx] = (__bf16)(v * pmv);   // gated half of concat
            }
        }
}

// ---------------------------------------------------------------------------
// GEMM 2: h = elu( G @ Wc[:, :512]^T + P @ Wc[:, 512:]^T + b_cat )
// 32 K-steps; A-source flips G -> P halfway.  (K = 1024)
// ---------------------------------------------------------------------------
__global__ void __launch_bounds__(256)
gemm_cat_kernel(const __bf16* __restrict__ P,
                const __bf16* __restrict__ G,
                const __bf16* __restrict__ Wcat,   // [512][1024] bf16
                const float* __restrict__ b_cat,
                __bf16* __restrict__ Hbuf) {
    __shared__ __attribute__((aligned(16))) __bf16 As[2][TN * ASTR];
    __shared__ __attribute__((aligned(16))) __bf16 Bs[2][TH * ASTR];

    int n0 = blockIdx.x * TN, h0 = blockIdx.y * TH, b = blockIdx.z;
    int tid = threadIdx.x, lane = tid & 31, wave = tid >> 5;
    int wn = wave & 1, wh = wave >> 1;

    int arow = tid >> 2, acol = (tid & 3) * 8;
    size_t aoff = ((size_t)b * NDIM + n0 + arow) * HID + acol;
    const __bf16* agpG = G + aoff;
    const __bf16* agpP = P + aoff;
    const __bf16* bgp  = Wcat + (size_t)(h0 + tid) * (2 * HID);

    auto issue = [&](int ks, int buf) {
        const __bf16* as = (ks < 16 ? agpG : agpP) + (ks & 15) * TK;
        const __bf16* bs = bgp + ks * TK;   // 2H-wide rows: ks*32 spans both halves
        async_cp16(as, &As[buf][arow * ASTR + acol]);
#pragma unroll
        for (int c = 0; c < 4; ++c)
            async_cp16(bs + c * 8, &Bs[buf][tid * ASTR + c * 8]);
        __builtin_prefetch(as + TK, 0, 3);
        __builtin_prefetch(bs + TK, 0, 3);
    };

    v8f acc[2][4] = {};
    constexpr int NSTEP = 2 * HID / TK;   // 32

    issue(0, 0);
#pragma unroll 2
    for (int ks = 0; ks < NSTEP; ++ks) {
        int cur = ks & 1;
        async_wait0();
        __syncthreads();
        if (ks + 1 < NSTEP) issue(ks + 1, cur ^ 1);
        wave_compute(As[cur], Bs[cur], wn, wh, lane, acc);
    }

    int m = lane & 15, hi = lane >> 4;
#pragma unroll
    for (int tn = 0; tn < 2; ++tn)
#pragma unroll
        for (int th = 0; th < 4; ++th) {
            int hh = h0 + wh * 64 + th * 16 + m;
            float bias = b_cat[hh];
#pragma unroll
            for (int r = 0; r < 8; ++r) {
                int nn = n0 + wn * 32 + tn * 16 + hi * 8 + r;
                float v = elu1(acc[tn][th][r] + bias);
                Hbuf[((size_t)b * NDIM + nn) * HID + hh] = (__bf16)v;
            }
        }
}

// ---------------------------------------------------------------------------
// GEMM 3: h2 = Hbuf @ Wcat2^T + b_cat2 ; attn = elu(h2 * ctrl) ;
//         logits[b,n] += sum_h attn * W_attn[h]   (partial over this h-block)
// ---------------------------------------------------------------------------
__global__ void __launch_bounds__(256)
gemm_attn_kernel(const __bf16* __restrict__ Hbuf,
                 const __bf16* __restrict__ Wcat2,
                 const float* __restrict__ b_cat2,
                 const float* __restrict__ control,
                 const float* __restrict__ W_attn,
                 float* __restrict__ logits) {
    __shared__ __attribute__((aligned(16))) __bf16 As[2][TN * ASTR];
    __shared__ __attribute__((aligned(16))) __bf16 Bs[2][TH * ASTR];

    int n0 = blockIdx.x * TN, h0 = blockIdx.y * TH, b = blockIdx.z;
    int tid = threadIdx.x, lane = tid & 31, wave = tid >> 5;
    int wn = wave & 1, wh = wave >> 1;

    int arow = tid >> 2, acol = (tid & 3) * 8;
    const __bf16* agp = Hbuf + ((size_t)b * NDIM + n0 + arow) * HID + acol;
    const __bf16* bgp = Wcat2 + (size_t)(h0 + tid) * HID;

    auto issue = [&](int ks, int buf) {
        const __bf16* as = agp + ks * TK;
        const __bf16* bs = bgp + ks * TK;
        async_cp16(as, &As[buf][arow * ASTR + acol]);
#pragma unroll
        for (int c = 0; c < 4; ++c)
            async_cp16(bs + c * 8, &Bs[buf][tid * ASTR + c * 8]);
        __builtin_prefetch(as + TK, 0, 3);
        __builtin_prefetch(bs + TK, 0, 3);
    };

    v8f acc[2][4] = {};
    constexpr int NSTEP = HID / TK;

    issue(0, 0);
#pragma unroll 2
    for (int ks = 0; ks < NSTEP; ++ks) {
        int cur = ks & 1;
        async_wait0();
        __syncthreads();
        if (ks + 1 < NSTEP) issue(ks + 1, cur ^ 1);
        wave_compute(As[cur], Bs[cur], wn, wh, lane, acc);
    }

    int m = lane & 15, hi = lane >> 4;
    const float* ctrl = control + ((size_t)(TDIM - 1) * BDIM + b) * HID;

    float cb[4], cv[4], wv[4];
#pragma unroll
    for (int th = 0; th < 4; ++th) {
        int hh = h0 + wh * 64 + th * 16 + m;
        cb[th] = b_cat2[hh];
        cv[th] = ctrl[hh];
        wv[th] = W_attn[hh];
    }

#pragma unroll
    for (int tn = 0; tn < 2; ++tn)
#pragma unroll
        for (int r = 0; r < 8; ++r) {
            float s = 0.0f;
#pragma unroll
            for (int th = 0; th < 4; ++th) {
                float v = acc[tn][th][r] + cb[th];
                v = elu1(v * cv[th]);
                s += v * wv[th];
            }
            // butterfly-reduce over the 16 columns (lanes share hi half)
            s += __shfl_xor(s, 1);
            s += __shfl_xor(s, 2);
            s += __shfl_xor(s, 4);
            s += __shfl_xor(s, 8);
            if (m == 0) {
                int nn = n0 + wn * 32 + tn * 16 + hi * 8 + r;
                atomicAdd(&logits[(size_t)b * NDIM + nn], s);
            }
        }
}

// ---------------------------------------------------------------------------
// Softmax over N per batch
// ---------------------------------------------------------------------------
__global__ void softmax_kernel(const float* __restrict__ logits,
                               float* __restrict__ aprob) {
    __shared__ float red[256];
    int b = blockIdx.x, tid = threadIdx.x;
    const float* L = logits + (size_t)b * NDIM;
    float* A = aprob + (size_t)b * NDIM;

    float mx = -3.4e38f;
    for (int n = tid; n < NDIM; n += 256) mx = fmaxf(mx, L[n]);
    red[tid] = mx;
    __syncthreads();
    for (int s = 128; s > 0; s >>= 1) {
        if (tid < s) red[tid] = fmaxf(red[tid], red[tid + s]);
        __syncthreads();
    }
    mx = red[0];
    __syncthreads();

    float sum = 0.0f;
    for (int n = tid; n < NDIM; n += 256) {
        float e = __expf(L[n] - mx);
        A[n] = e;
        sum += e;
    }
    red[tid] = sum;
    __syncthreads();
    for (int s = 128; s > 0; s >>= 1) {
        if (tid < s) red[tid] += red[tid + s];
        __syncthreads();
    }
    float inv = 1.0f / red[0];
    __syncthreads();
    for (int n = tid; n < NDIM; n += 256) A[n] *= inv;
}

// ---------------------------------------------------------------------------
// read[b,kk] = sum_n a[b,n] * know[b,kk,n]   (memory-bound GEMV on f32 know)
// ---------------------------------------------------------------------------
__global__ void read_kernel(const float* __restrict__ know,
                            const float* __restrict__ aprob,
                            float* __restrict__ out) {
    __shared__ float red[256];
    int idx = blockIdx.x, tid = threadIdx.x;
    int b = idx / KBDIM, kk = idx % KBDIM;
    const float* row = know + ((size_t)b * KBDIM + kk) * NDIM;
    const float* A = aprob + (size_t)b * NDIM;
    float s = 0.0f;
    for (int n = tid; n < NDIM; n += 256) s += A[n] * row[n];
    red[tid] = s;
    __syncthreads();
    for (int t = 128; t > 0; t >>= 1) {
        if (tid < t) red[tid] += red[tid + t];
        __syncthreads();
    }
    if (tid == 0) out[idx] = red[0];
}

// ---------------------------------------------------------------------------
// Host launcher
// ---------------------------------------------------------------------------
extern "C" void kernel_launch(void* const* d_in, const int* in_sizes, int n_in,
                              void* d_out, int out_size, void* d_ws, size_t ws_size,
                              hipStream_t stream) {
    const float* memory  = (const float*)d_in[0];
    const float* know    = (const float*)d_in[1];
    const float* control = (const float*)d_in[2];
    const float* masks   = (const float*)d_in[3];
    const float* W_mem   = (const float*)d_in[4];
    const float* b_mem   = (const float*)d_in[5];
    const float* W_kb    = (const float*)d_in[6];
    const float* b_kb    = (const float*)d_in[7];
    const float* W_cat   = (const float*)d_in[8];
    const float* b_cat   = (const float*)d_in[9];
    const float* W_cat2  = (const float*)d_in[10];
    const float* b_cat2  = (const float*)d_in[11];
    const float* W_attn  = (const float*)d_in[12];
    // d_in[13] = b_attn: uniform logit shift, softmax-invariant -> dropped.

    char* ws = (char*)d_ws;
    constexpr size_t SZ_PM     = (size_t)BDIM * HID * 4;
    constexpr size_t SZ_LOG    = (size_t)BDIM * NDIM * 4;
    constexpr size_t SZ_WKB    = (size_t)HID * KBDIM * 2;
    constexpr size_t SZ_WCAT   = (size_t)HID * 2 * HID * 2;
    constexpr size_t SZ_WCAT2  = (size_t)HID * HID * 2;
    constexpr size_t SZ_KT     = (size_t)BDIM * NDIM * KBDIM * 2;

    size_t off = 0;
    float*  pm      = (float*)(ws + off); off += SZ_PM;
    float*  logits  = (float*)(ws + off); off += SZ_LOG;
    float*  aprob   = (float*)(ws + off); off += SZ_LOG;
    __bf16* Wkb_bf  = (__bf16*)(ws + off); off += SZ_WKB;
    __bf16* Wcat_bf = (__bf16*)(ws + off); off += SZ_WCAT;
    __bf16* Wc2_bf  = (__bf16*)(ws + off); off += SZ_WCAT2;
    __bf16* kT      = (__bf16*)(ws + off); off += SZ_KT;
    __bf16* Pbuf    = (__bf16*)(ws + off); off += SZ_KT;
    __bf16* Gbuf    = (__bf16*)(ws + off); off += SZ_KT;
    __bf16* Hbuf    = (__bf16*)(ws + off); off += SZ_KT;

    projmem_kernel<<<dim3(HID / 256, BDIM), 256, 0, stream>>>(memory, masks, W_mem, b_mem, pm);

    cvt_bf16_kernel<<<(HID * KBDIM + 255) / 256, 256, 0, stream>>>(W_kb, Wkb_bf, HID * KBDIM);
    cvt_bf16_kernel<<<(HID * 2 * HID + 255) / 256, 256, 0, stream>>>(W_cat, Wcat_bf, HID * 2 * HID);
    cvt_bf16_kernel<<<(HID * HID + 255) / 256, 256, 0, stream>>>(W_cat2, Wc2_bf, HID * HID);

    transpose_know_kernel<<<dim3(NDIM / 32, KBDIM / 32, BDIM), dim3(32, 8), 0, stream>>>(know, kT);

    zero_f32_kernel<<<(BDIM * NDIM + 255) / 256, 256, 0, stream>>>(logits, BDIM * NDIM);

    dim3 ggrid(NDIM / TN, HID / TH, BDIM);

    gemm_projknow_kernel<<<ggrid, 256, 0, stream>>>(kT, Wkb_bf, b_kb, pm, Pbuf, Gbuf);
    gemm_cat_kernel<<<ggrid, 256, 0, stream>>>(Pbuf, Gbuf, Wcat_bf, b_cat, Hbuf);
    gemm_attn_kernel<<<ggrid, 256, 0, stream>>>(Hbuf, Wc2_bf, b_cat2, control, W_attn, logits);

    softmax_kernel<<<BDIM, 256, 0, stream>>>(logits, aprob);
    read_kernel<<<BDIM * KBDIM, 256, 0, stream>>>(know, aprob, (float*)d_out);
}
